// Net_15668040696431
// MI455X (gfx1250) — matrix operand order repaired and checked by reading, not compiled, not run on hardware
//
#include <hip/hip_runtime.h>

typedef unsigned short u16;
typedef __attribute__((ext_vector_type(16))) __bf16 bf16x16;
typedef __attribute__((ext_vector_type(8)))  float  floatx8;
typedef __attribute__((ext_vector_type(4))) unsigned int uintx4;
typedef __attribute__((ext_vector_type(8))) int intx8;
typedef __attribute__((ext_vector_type(4))) int intx4;

#define BM 64
#define BN 64
#define BK 32

__device__ __forceinline__ u16 f32_to_bf16_rne(float f) {
  unsigned u = __float_as_uint(f);
  u += 0x7FFFu + ((u >> 16) & 1u);   // round-to-nearest-even
  return (u16)(u >> 16);
}

union Frag {
  bf16x16 v;
  uint4   q[2];
};

struct GemmP {
  const u16*   A;     // [M, lda] bf16 row-major
  const u16*   B;     // [N, ldb] bf16 = math-B transposed (row n = column n), batched via sB
  const float* U;     // optional fp32 addend, [.., ldu] row-major, batched via sU
  const float* bias;  // optional [N]
  float*       outF;  // mode 0
  u16*         outH;  // mode 1 / 2
  int M, N, K;
  int lda, ldb, ldu, ldo;
  long long sB, sU, sO;  // batch strides (elements)
  int mode;              // 0=f32 row-major, 1=bf16 row-major, 2=bf16 batch-transposed
  int relu;
  int rowsPerBatch;      // mode 2: rows per logical batch (400)
};

// Padded LDS layout: 16B pad after every 32 bf16 elements -> row stride (K/32)*80 B.
__device__ __forceinline__ unsigned lds_off(int row, int k, unsigned rs) {
  return (unsigned)row * rs + ((unsigned)k << 1) + (((unsigned)k >> 5) << 4);
}

#if __has_builtin(__builtin_amdgcn_tensor_load_to_lds) && __has_builtin(__builtin_amdgcn_s_wait_tensorcnt)
#define USE_TDM 1
// One TDM op: load a [rows<=64] x [tile_k] bf16 2-D tile (row stride lda elems) into LDS
// at byte offset lds_base, with 16B LDS padding every 64B stored (matches lds_off layout).
// tensor_dim = remaining extent from the tile start => OOB rows/cols are zero-filled.
__device__ __forceinline__ void tdm_load_2d(unsigned lds_base, const u16* gptr,
                                            int k_rem, int rows_rem, int tile_k, int lda) {
  unsigned long long ga = (unsigned long long)(uintptr_t)gptr;
  uintx4 g0;
  g0.x = 1u;                                            // count=1, user mode, no gather
  g0.y = lds_base;                                      // lds_addr (bytes)
  g0.z = (unsigned)(ga & 0xFFFFFFFFull);                // global_addr[31:0]
  g0.w = (unsigned)((ga >> 32) & 0x01FFFFFFull) | (2u << 30);  // addr[56:32] | type=2
  intx8 g1;
  // data_size=1 (2B) | pad_enable | pad_interval=3 (16 DW = 64B) | pad_amount=3 (4 DW = 16B)
  g1[0] = (int)((1u << 16) | (1u << 20) | (3u << 22) | (3u << 25));
  g1[1] = (int)(((unsigned)k_rem & 0xFFFFu) << 16);                       // tensor_dim0 lo
  g1[2] = (int)((((unsigned)k_rem >> 16) & 0xFFFFu)
                | (((unsigned)rows_rem & 0xFFFFu) << 16));                // dim0 hi | dim1 lo
  g1[3] = (int)((((unsigned)rows_rem >> 16) & 0xFFFFu)
                | (((unsigned)tile_k & 0xFFFFu) << 16));                  // dim1 hi | tile_dim0
  g1[4] = 64;                                           // tile_dim1=64 rows, tile_dim2=0
  g1[5] = lda;                                          // tensor_dim0_stride (elements) lo32
  g1[6] = 0;                                            // stride0 hi | stride1 lo
  g1[7] = 0;
  intx4 gz4 = {0, 0, 0, 0};
  intx8 gz8 = {0, 0, 0, 0, 0, 0, 0, 0};
  // 6-arg toolchain form: (g0, g1, g2, g3, g4, cpol)
  __builtin_amdgcn_tensor_load_to_lds(g0, g1, gz4, gz4, gz8, 0);
}
#else
#define USE_TDM 0
#endif

__global__ __launch_bounds__(128)
void gemm_bf16_wmma(GemmP p) {
  extern __shared__ __align__(16) char dynsmem[];

  const int tid  = threadIdx.x;
  const int lane = tid & 31;
  const int wave = tid >> 5;
  const int wr   = wave >> 1;     // 2x2 wave grid over 64x64 tile
  const int wc   = wave & 1;
  const int lrow = lane & 15;
  const int half = lane >> 4;

  const int m0 = blockIdx.y * BM;
  const int n0 = blockIdx.x * BN;
  const int z  = blockIdx.z;

  const u16* Ag = p.A;
  const u16* Bg = p.B + (size_t)z * (size_t)p.sB;

  const unsigned rs = (unsigned)(p.K >> 5) * 80u;   // padded LDS row stride (bytes)
  char* As = dynsmem;                                // A strip: 64 rows x K
  char* Bs = dynsmem + (size_t)64u * rs;             // B strip: 64 rows x K

#if USE_TDM
  if (wave == 0) {
    unsigned baseL = (unsigned)(uintptr_t)dynsmem;
    tdm_load_2d(baseL,            Ag + (size_t)m0 * p.lda, p.K, p.M - m0, p.K, p.lda);
    tdm_load_2d(baseL + 64u * rs, Bg + (size_t)n0 * p.ldb, p.K, p.N - n0, p.K, p.ldb);
    __builtin_amdgcn_s_wait_tensorcnt(0);
  }
  __syncthreads();
#else
  {
    const int kchunks = p.K >> 3;             // 8-element (16B) chunks per row
    const int total   = 64 * kchunks;
    for (int c = tid; c < total; c += 128) {
      int row = c / kchunks;
      int kc  = (c - row * kchunks) << 3;
      uint4 av = make_uint4(0u, 0u, 0u, 0u);
      if (m0 + row < p.M) av = *(const uint4*)(Ag + (size_t)(m0 + row) * p.lda + kc);
      *(uint4*)(As + lds_off(row, kc, rs)) = av;
      uint4 bv = make_uint4(0u, 0u, 0u, 0u);
      if (n0 + row < p.N) bv = *(const uint4*)(Bg + (size_t)(n0 + row) * p.ldb + kc);
      *(uint4*)(Bs + lds_off(row, kc, rs)) = bv;
    }
  }
  __syncthreads();
#endif

  floatx8 acc[2][2];
#pragma unroll
  for (int i = 0; i < 2; ++i)
#pragma unroll
    for (int j = 0; j < 2; ++j)
      acc[i][j] = (floatx8){0.f,0.f,0.f,0.f,0.f,0.f,0.f,0.f};

  // Barrier-free hot loop: pure ds_load_b128 + v_wmma.
  for (int kk = 0; kk < p.K; kk += BK) {
    Frag fa[2], fb[2];
#pragma unroll
    for (int mi = 0; mi < 2; ++mi) {
      int r = wr * 32 + mi * 16 + lrow;
      // 16-bit A layout: elems 0..7 <- K = half*8..+7 ; elems 8..15 <- K = 16+half*8..+7
      fa[mi].q[0] = *(const uint4*)(As + lds_off(r, kk + half * 8, rs));
      fa[mi].q[1] = *(const uint4*)(As + lds_off(r, kk + 16 + half * 8, rs));
    }
#pragma unroll
    for (int ni = 0; ni < 2; ++ni) {
      int c = wc * 32 + ni * 16 + lrow;
      // 16-bit B layout: elems 0..7 <- K = half*8..+7 ; elems 8..15 <- K = 16+half*8..+7
      fb[ni].q[0] = *(const uint4*)(Bs + lds_off(c, kk + half * 8, rs));
      fb[ni].q[1] = *(const uint4*)(Bs + lds_off(c, kk + 16 + half * 8, rs));
    }
#pragma unroll
    for (int mi = 0; mi < 2; ++mi)
#pragma unroll
      for (int ni = 0; ni < 2; ++ni)
        acc[mi][ni] = __builtin_amdgcn_wmma_f32_16x16x32_bf16(
            false, fa[mi].v, false, fb[ni].v,
            (short)0, acc[mi][ni], false, false);
  }

  const float* Ub = p.U ? (p.U + (size_t)z * (size_t)p.sU) : (const float*)0;

#pragma unroll
  for (int mi = 0; mi < 2; ++mi) {
#pragma unroll
    for (int ni = 0; ni < 2; ++ni) {
      int col = n0 + wc * 32 + ni * 16 + lrow;      // D: N = lane%16
      if (col >= p.N) continue;
      float bv = p.bias ? p.bias[col] : 0.f;
#pragma unroll
      for (int e = 0; e < 8; ++e) {
        int row = m0 + wr * 32 + mi * 16 + half * 8 + e;   // D: M = vgpr + 8*(lane/16)
        if (row >= p.M) continue;
        float d = acc[mi][ni][e] + bv;
        if (Ub) d += Ub[(size_t)row * p.ldu + col];
        if (p.relu) d = d > 0.f ? d : 0.f;   // elu(relu(x)) == relu(x)
        if (p.mode == 0) {
          p.outF[(size_t)z * (size_t)p.sO + (size_t)row * p.ldo + col] = d;
        } else if (p.mode == 1) {
          p.outH[(size_t)z * (size_t)p.sO + (size_t)row * p.ldo + col] = f32_to_bf16_rne(d);
        } else {
          int b = row / p.rowsPerBatch;
          int j = row - b * p.rowsPerBatch;
          p.outH[(size_t)b * (size_t)p.sO + (size_t)col * p.ldo + j] = f32_to_bf16_rne(d);
        }
      }
    }
  }
}

// fp32 [rows, cols] -> bf16 [rows, ldout] (pad columns pre-zeroed by memset)
__global__ void cvt_rm(const float* in, u16* out, long long total, int cols, int ldout) {
  long long i = (long long)blockIdx.x * blockDim.x + threadIdx.x;
  long long stride = (long long)gridDim.x * blockDim.x;
  for (; i < total; i += stride) {
    int r = (int)(i / cols);
    int c = (int)(i - (long long)r * cols);
    out[(size_t)r * ldout + c] = f32_to_bf16_rne(in[i]);
  }
}

// fp32 [K, ncols] row-major -> bf16 transposed [ncols, ldout]
__global__ void cvt_tr(const float* in, u16* out, long long total, int ncols, int ldout) {
  long long i = (long long)blockIdx.x * blockDim.x + threadIdx.x;
  long long stride = (long long)gridDim.x * blockDim.x;
  for (; i < total; i += stride) {
    int k = (int)(i / ncols);
    int n = (int)(i - (long long)k * ncols);
    out[(size_t)n * ldout + k] = f32_to_bf16_rne(in[i]);
  }
}

static inline void launch_gemm(const GemmP& p, int batches, hipStream_t stream) {
  size_t shbytes = (size_t)128u * (size_t)(p.K >> 5) * 80u + 16u;  // A+B strips, padded
  dim3 grid((unsigned)(p.N / BN), (unsigned)((p.M + BM - 1) / BM), (unsigned)batches);
  hipLaunchKernelGGL(gemm_bf16_wmma, grid, dim3(128), (unsigned)shbytes, stream, p);
}

static inline int cvt_blocks(long long total) {
  long long b = (total + 255) / 256;
  return (int)(b < 4096 ? b : 4096);
}

extern "C" void kernel_launch(void* const* d_in, const int* in_sizes, int n_in,
                              void* d_out, int out_size, void* d_ws, size_t ws_size,
                              hipStream_t stream) {
  (void)in_sizes; (void)n_in; (void)out_size; (void)ws_size;
  const float* x  = (const float*)d_in[0];   // [64,400,240]
  const float* a  = (const float*)d_in[1];   // [400,400]
  const float* w1 = (const float*)d_in[2];   // [240,128]
  const float* v1 = (const float*)d_in[3];
  const float* b1 = (const float*)d_in[4];   // [128]
  const float* w2 = (const float*)d_in[5];   // [128,128]
  const float* v2 = (const float*)d_in[6];
  const float* b2 = (const float*)d_in[7];
  const float* wd = (const float*)d_in[8];   // [128,2880]
  const float* bd = (const float*)d_in[9];   // [2880]
  float* out = (float*)d_out;                // [64,400,2880]

  const int Bb = 64, Nn = 400, FIN = 240, FINP = 256, C = 128, NL = 2880, NP = 416;
  const int MR = Bb * Nn;  // 25600

  char* ws = (char*)d_ws;
  size_t off = 0;
  auto alloc = [&](size_t bytes) -> void* {
    void* p = ws + off;
    off = (off + bytes + 255) & ~(size_t)255;
    return p;
  };
  u16*   Xb  = (u16*)alloc((size_t)MR * FINP * 2);      // x, K-padded
  u16*   Ab  = (u16*)alloc((size_t)Nn * NP * 2);        // A, K-padded
  u16*   W1t = (u16*)alloc((size_t)C * FINP * 2);
  u16*   V1t = (u16*)alloc((size_t)C * FINP * 2);
  u16*   W2t = (u16*)alloc((size_t)C * C * 2);
  u16*   V2t = (u16*)alloc((size_t)C * C * 2);
  u16*   WDt = (u16*)alloc((size_t)NL * C * 2);
  u16*   Tt  = (u16*)alloc((size_t)Bb * C * NP * 2);    // T transposed per batch [b][c][j<=416]
  float* Uf  = (float*)alloc((size_t)Bb * Nn * C * 4);  // xV residual
  u16*   H   = (u16*)alloc((size_t)MR * C * 2);         // layer activations (bf16)

  // zero workspace: provides the zero K/j padding everywhere
  (void)hipMemsetAsync(d_ws, 0, off, stream);

  // --- stage bf16 operands ---
  hipLaunchKernelGGL(cvt_rm, dim3(cvt_blocks((long long)MR * FIN)), dim3(256), 0, stream,
                     x, Xb, (long long)MR * FIN, FIN, FINP);
  hipLaunchKernelGGL(cvt_rm, dim3(cvt_blocks((long long)Nn * Nn)), dim3(256), 0, stream,
                     a, Ab, (long long)Nn * Nn, Nn, NP);
  hipLaunchKernelGGL(cvt_tr, dim3(cvt_blocks((long long)FIN * C)), dim3(256), 0, stream,
                     w1, W1t, (long long)FIN * C, C, FINP);
  hipLaunchKernelGGL(cvt_tr, dim3(cvt_blocks((long long)FIN * C)), dim3(256), 0, stream,
                     v1, V1t, (long long)FIN * C, C, FINP);
  hipLaunchKernelGGL(cvt_tr, dim3(cvt_blocks((long long)C * C)), dim3(256), 0, stream,
                     w2, W2t, (long long)C * C, C, C);
  hipLaunchKernelGGL(cvt_tr, dim3(cvt_blocks((long long)C * C)), dim3(256), 0, stream,
                     v2, V2t, (long long)C * C, C, C);
  hipLaunchKernelGGL(cvt_tr, dim3(cvt_blocks((long long)C * NL)), dim3(256), 0, stream,
                     wd, WDt, (long long)C * NL, NL, C);

  // --- three ARMA conv layers (layers 2 and 3 reuse w2/v2/b2 per the reference) ---
  for (int layer = 0; layer < 3; ++layer) {
    const u16* act  = (layer == 0) ? Xb  : H;
    const u16* Wt   = (layer == 0) ? W1t : W2t;
    const u16* Vt   = (layer == 0) ? V1t : V2t;
    const float* bs = (layer == 0) ? b1  : b2;
    int Kl          = (layer == 0) ? FINP : C;

    // T[b,j,c] = act @ W, stored transposed bf16 into Tt[b][c][j]
    GemmP pT{};
    pT.A = act; pT.B = Wt; pT.M = MR; pT.N = C; pT.K = Kl;
    pT.lda = Kl; pT.ldb = Kl;
    pT.mode = 2; pT.outH = Tt; pT.ldo = NP; pT.sO = (long long)C * NP;
    pT.rowsPerBatch = Nn;
    launch_gemm(pT, 1, stream);

    // U[b,i,c] = act @ V, fp32
    GemmP pU{};
    pU.A = act; pU.B = Vt; pU.M = MR; pU.N = C; pU.K = Kl;
    pU.lda = Kl; pU.ldb = Kl;
    pU.mode = 0; pU.outF = Uf; pU.ldo = C;
    launch_gemm(pU, 1, stream);

    // H[b] = relu(A @ T[b] + U[b] + bias), batched over grid.z, bf16 out
    GemmP pH{};
    pH.A = Ab; pH.B = Tt; pH.M = Nn; pH.N = C; pH.K = NP;
    pH.lda = NP; pH.ldb = NP; pH.sB = (long long)C * NP;
    pH.U = Uf; pH.ldu = C; pH.sU = (long long)Nn * C;
    pH.bias = bs; pH.relu = 1;
    pH.mode = 1; pH.outH = H; pH.ldo = C; pH.sO = (long long)Nn * C;
    launch_gemm(pH, Bb, stream);
  }

  // --- dense head: out = H @ Wd + bd, fp32 ---
  GemmP pD{};
  pD.A = H; pD.B = WDt; pD.M = MR; pD.N = NL; pD.K = C;
  pD.lda = C; pD.ldb = C;
  pD.bias = bd;
  pD.mode = 0; pD.outF = out; pD.ldo = NL;
  launch_gemm(pD, 1, stream);
}
